// MultiHeadSelfAtt_65317862637786
// MI455X (gfx1250) — matrix operand order repaired and checked by reading, compiled, not run on hardware
//
#include <hip/hip_runtime.h>
#include <hip/hip_bf16.h>

typedef __attribute__((ext_vector_type(16))) _Float16 v16h;
typedef __attribute__((ext_vector_type(8)))  _Float16 v8h;
typedef __attribute__((ext_vector_type(8)))  float    v8f;
typedef __attribute__((ext_vector_type(4)))  int      v4i;

#define LTOK 4096   // H*W
#define CTOT 256
#define HEADS 8
#define CPH 32
#define JT 64       // attention j-tile

#if __has_builtin(__builtin_amdgcn_global_load_async_to_lds_b128)
#define USE_ASYNC 1
#else
#define USE_ASYNC 0
#endif

#if USE_ASYNC
#if __has_builtin(__builtin_amdgcn_s_wait_asynccnt)
#define WAIT_ASYNC(n) __builtin_amdgcn_s_wait_asynccnt(n)
#else
#define WAIT_ASYNC(n) asm volatile("s_wait_asynccnt %0" :: "i"(n) : "memory")
#endif
// builtin params are int4* in AS1 (global) and AS3 (LDS); cast via integer reinterpret
#define AS1P(p) ((__attribute__((address_space(1))) v4i*)(unsigned long long)(p))
#define AS3P(p) ((__attribute__((address_space(3))) v4i*)(unsigned int)(unsigned long long)(p))
#else
#define WAIT_ASYNC(n) ((void)0)
#endif

__device__ __forceinline__ v16h cat16(v8h a, v8h b) {
  return __builtin_shufflevector(a, b, 0,1,2,3,4,5,6,7,8,9,10,11,12,13,14,15);
}
// A-operand (16x32 f16): lane row m = lane&15; K = 2*kb + e (kb = 8*(lane>>4))
__device__ __forceinline__ v16h loadA(const _Float16* rowbase, int kb) {
  v8h lo = *(const v8h*)(rowbase + 2 * kb);
  v8h hi = *(const v8h*)(rowbase + 2 * kb + 8);
  return cat16(lo, hi);
}
// B-operand (32x16 f16): lane col n = lane&15; K = 16*(lane>>4) + e (one 32B run)
__device__ __forceinline__ v16h loadB(const _Float16* p) {
  v8h lo = *(const v8h*)(p);
  v8h hi = *(const v8h*)(p + 8);
  return cat16(lo, hi);
}
__device__ __forceinline__ v8f wmma16(v16h a, v16h b, v8f c) {
  return __builtin_amdgcn_wmma_f32_16x16x32_f16(false, a, false, b, (short)0, c, false, false);
}

// ---------------- Stage 0a: weights f32 -> f16 (wq|wk|wv|wl) -----------------
__global__ void prep_w_kernel(const float* __restrict__ wq, const float* __restrict__ wk,
                              const float* __restrict__ wv, const float* __restrict__ wl,
                              _Float16* __restrict__ wf) {
  int i = blockIdx.x * 256 + threadIdx.x;            // 4*65536 total
  int m = i >> 16, e = i & 65535;
  float v;
  if      (m == 0) v = wq[e];
  else if (m == 1) v = wk[e];
  else if (m == 2) v = wv[e];
  else             v = wl[e];
  wf[i] = (_Float16)v;
}

// ---------------- Stage 0b: x[n][c][l] f32 -> xT[n][l][c] f16 ----------------
__global__ void prep_x_kernel(const float* __restrict__ x, _Float16* __restrict__ xT) {
  int i = blockIdx.x * 256 + threadIdx.x;            // 2*256*4096 total
  int c = i & 255;
  int l = (i >> 8) & 4095;
  int n = i >> 20;
  xT[(((size_t)n * LTOK + l) * CTOT) + c] = (_Float16)x[((size_t)n * CTOT + c) * LTOK + l];
}

// ---------------- Stage 1: fused q/k/v projection (WMMA GEMM) ----------------
__global__ void proj_kernel(const _Float16* __restrict__ xT,  // [n][l][256]
                            const _Float16* __restrict__ wf,  // wq|wk|wv f16 [256][256]
                            const float* __restrict__ bq, const float* __restrict__ bk,
                            const float* __restrict__ bv,
                            _Float16* __restrict__ qT,        // [nh][l][32]
                            _Float16* __restrict__ kT,        // [nh][i][32]
                            _Float16* __restrict__ vO) {      // [n][c][L]
  int t    = blockIdx.x * 8 + (threadIdx.x >> 5);
  int lane = threadIdx.x & 31;
  int half = lane >> 4, idx = lane & 15, kb = half * 8;
  int l0 = (t & 255) * 16;
  int o0 = ((t >> 8) & 15) * 16;
  int n  = t >> 12;

  v8f aq = {}, ak = {}, av = {};
  const _Float16* xrow = xT + ((size_t)n * LTOK + l0 + idx) * CTOT;
  const _Float16* wqr  = wf + (size_t)(o0 + idx) * CTOT;
  const _Float16* wkr  = wqr + 65536;
  const _Float16* wvr  = wqr + 131072;
  #pragma unroll
  for (int c0 = 0; c0 < CTOT; c0 += 32) {
    v16h bx = loadB(xrow + c0 + half * 16);
    aq = wmma16(loadA(wqr + c0, kb), bx, aq);
    ak = wmma16(loadA(wkr + c0, kb), bx, ak);
    av = wmma16(loadA(wvr + c0, kb), bx, av);
  }
  int l = l0 + idx;
  #pragma unroll
  for (int r = 0; r < 8; r++) {
    int o = o0 + r + 8 * half;
    int h = o >> 5, d = o & 31;
    size_t nh = (size_t)n * HEADS + h;
    qT[(nh * LTOK + l) * CPH + d] = (_Float16)(aq[r] + bq[o]);
    kT[(nh * LTOK + l) * CPH + d] = (_Float16)(ak[r] + bk[o]);
    vO[((size_t)n * CTOT + o) * LTOK + l] = (_Float16)(av[r] + bv[o]);
  }
}

// ---------------- Stage 2: flash attention -----------------------------------
// att[i,j] = softmax_j( k[:,i].q[:,j] * scale );  O^T[i,d] = sum_j att[i,j] v[d,j]
// One wave owns 16 i-rows; 8 waves/WG; 32 WGs per (n,h); j streamed in tiles of 64
// staged into LDS via async global->LDS (double buffered).
__global__ void attn_kernel(const _Float16* __restrict__ kT,   // [nh][i][32]
                            const _Float16* __restrict__ qT,   // [nh][j][32]
                            const _Float16* __restrict__ v,    // [n][c][L]
                            _Float16* __restrict__ attnT) {    // [n][l][256]
  __shared__ _Float16 qs[2][JT * 40];      // Q tile [j][32] padded to 40 (16B-aligned rows)
  __shared__ _Float16 vs[2][CPH * 72];     // V tile [d][64] padded to 72
  __shared__ _Float16 pslab[8][16 * 72];   // per-wave P re-layout slab [i][64] pad 72

  const float SCL2 = 0.17677669529663687f * 1.4426950408889634f;  // cph^-0.5 * log2(e)
  int wg = blockIdx.x, nh = wg >> 5, ib = wg & 31;
  int n = nh >> 3, h = nh & 7;
  int tid = threadIdx.x;
  int wave = tid >> 5, lane = tid & 31;
  int half = lane >> 4, idx = lane & 15, kb = half * 8;
  int i0 = ib * 128 + wave * 16;

  const _Float16* qbase = qT + (size_t)nh * LTOK * CPH;              // [j][32] contiguous
  const _Float16* vbase = v + ((size_t)n * CTOT + h * CPH) * LTOK;   // [d][L]

  // cooperative staging roles (256 threads)
  int qrow = tid >> 2, qseg = tid & 3;     // Q: 64 rows x 4 x 16B
  int vrow = tid >> 3, vseg = tid & 7;     // V: 32 rows x 8 x 16B

  auto stage = [&](int buf, int j0) {
    const _Float16* gq = qbase + (size_t)(j0 + qrow) * CPH + qseg * 8;
    _Float16* lq = &qs[buf][qrow * 40 + qseg * 8];
    const _Float16* gv = vbase + (size_t)vrow * LTOK + j0 + vseg * 8;
    _Float16* lv = &vs[buf][vrow * 72 + vseg * 8];
#if USE_ASYNC
    __builtin_amdgcn_global_load_async_to_lds_b128(AS1P(gq), AS3P(lq), 0, 0);
    __builtin_amdgcn_global_load_async_to_lds_b128(AS1P(gv), AS3P(lv), 0, 0);
#else
    *(v8h*)lq = *(const v8h*)gq;
    *(v8h*)lv = *(const v8h*)gv;
#endif
  };

  // A operand: K^T rows for this wave's 16 i's (held in registers all along)
  v16h aK = loadA(kT + ((size_t)nh * LTOK + i0 + idx) * CPH, kb);

  v8f o0v = {}, o1v = {};
  float mrun[8], lrun[8];
  #pragma unroll
  for (int r = 0; r < 8; r++) { mrun[r] = -1e30f; lrun[r] = 0.f; }
  _Float16* myp = &pslab[wave][0];

  auto compute = [&](int buf) {
    const _Float16* lq = &qs[buf][0];
    const _Float16* lv = &vs[buf][0];
    v8f s[4]; v8f zero = {};
    #pragma unroll
    for (int jt = 0; jt < 4; jt++) {
      v16h bq = loadB(lq + (jt * 16 + idx) * 40 + half * 16);
      s[jt] = wmma16(aK, bq, zero);
    }
    // online softmax: row m = r + 8*half lives within this lane's 16-lane half
    float alpha[8];
    #pragma unroll
    for (int r = 0; r < 8; r++) {
      float t0 = s[0][r] * SCL2, t1 = s[1][r] * SCL2;
      float t2 = s[2][r] * SCL2, t3 = s[3][r] * SCL2;
      float mx = fmaxf(fmaxf(t0, t1), fmaxf(t2, t3));
      #pragma unroll
      for (int off = 1; off < 16; off <<= 1) mx = fmaxf(mx, __shfl_xor(mx, off, 32));
      float mnew = fmaxf(mrun[r], mx);
      float p0 = exp2f(t0 - mnew), p1 = exp2f(t1 - mnew);
      float p2 = exp2f(t2 - mnew), p3 = exp2f(t3 - mnew);
      float a  = exp2f(mrun[r] - mnew);
      float ps = (p0 + p1) + (p2 + p3);
      #pragma unroll
      for (int off = 1; off < 16; off <<= 1) ps += __shfl_xor(ps, off, 32);
      lrun[r] = lrun[r] * a + ps;
      mrun[r] = mnew;
      alpha[r] = a;
      s[0][r] = p0; s[1][r] = p1; s[2][r] = p2; s[3][r] = p3;
    }
    #pragma unroll
    for (int r = 0; r < 8; r++) { o0v[r] *= alpha[r]; o1v[r] *= alpha[r]; }

    // P: f32 D-layout -> f16 A-layout via wave-private LDS (DS in-order per wave)
    #pragma unroll
    for (int jt = 0; jt < 4; jt++) {
      #pragma unroll
      for (int r = 0; r < 8; r++) {
        int m = r + 8 * half;
        myp[m * 72 + jt * 16 + idx] = (_Float16)s[jt][r];
      }
    }
    v16h aP0 = loadA(myp + idx * 72, kb);        // K = j 0..31
    v16h aP1 = loadA(myp + idx * 72 + 32, kb);   // K = j 32..63

    v16h bv00 = loadB(lv + idx * 72 + half * 16);             // d 0..15,  j 0..31
    v16h bv01 = loadB(lv + idx * 72 + 32 + half * 16);        // d 0..15,  j 32..63
    v16h bv10 = loadB(lv + (16 + idx) * 72 + half * 16);      // d 16..31, j 0..31
    v16h bv11 = loadB(lv + (16 + idx) * 72 + 32 + half * 16); // d 16..31, j 32..63
    o0v = wmma16(aP0, bv00, o0v);
    o0v = wmma16(aP1, bv01, o0v);
    o1v = wmma16(aP0, bv10, o1v);
    o1v = wmma16(aP1, bv11, o1v);
  };

  stage(0, 0);
  for (int it = 0; it < 63; ++it) {
    stage((it + 1) & 1, (it + 1) * JT);   // prefetch next tile into other buffer
    WAIT_ASYNC(2);                        // my current-tile copies done (in-order)
    __syncthreads();                      // everyone's current tile resident
    compute(it & 1);
    __syncthreads();                      // all readers done before buffer reuse
  }
  WAIT_ASYNC(0);
  __syncthreads();
  compute(1);                             // tile 63 lives in buffer 1

  // normalize, store attn^T as [n][l][c] f16 (B-operand-ready for output proj)
  #pragma unroll
  for (int r = 0; r < 8; r++) {
    int i = i0 + r + 8 * half;
    float inv = 1.0f / lrun[r];
    size_t base = ((size_t)n * LTOK + i) * CTOT + h * CPH;
    attnT[base + idx]      = (_Float16)(o0v[r] * inv);
    attnT[base + 16 + idx] = (_Float16)(o1v[r] * inv);
  }
}

// ---------------- Stage 3: output projection + gated residual ----------------
__global__ void outproj_kernel(const _Float16* __restrict__ attnT, // [n][l][256]
                               const _Float16* __restrict__ wlf,   // [256][256] f16
                               const float* __restrict__ bl, const float* __restrict__ x,
                               const float* __restrict__ gamma, float* __restrict__ out) {
  int t    = blockIdx.x * 8 + (threadIdx.x >> 5);
  int lane = threadIdx.x & 31;
  int half = lane >> 4, idx = lane & 15, kb = half * 8;
  int l0 = (t & 255) * 16;
  int o0 = ((t >> 8) & 15) * 16;
  int n  = t >> 12;

  v8f acc = {};
  const _Float16* arow = attnT + ((size_t)n * LTOK + l0 + idx) * CTOT;
  const _Float16* wrow = wlf + (size_t)(o0 + idx) * CTOT;
  #pragma unroll
  for (int c0 = 0; c0 < CTOT; c0 += 32) {
    v16h bx = loadB(arow + c0 + half * 16);
    acc = wmma16(loadA(wrow + c0, kb), bx, acc);
  }
  float g = gamma[0];
  int l = l0 + idx;
  #pragma unroll
  for (int r = 0; r < 8; r++) {
    int o = o0 + r + 8 * half;
    size_t off = ((size_t)n * CTOT + o) * LTOK + l;
    out[off] = g * (acc[r] + bl[o]) + x[off];
  }
}

extern "C" void kernel_launch(void* const* d_in, const int* in_sizes, int n_in,
                              void* d_out, int out_size, void* d_ws, size_t ws_size,
                              hipStream_t stream) {
  const float* x  = (const float*)d_in[0];
  const float* wq = (const float*)d_in[1];
  const float* bq = (const float*)d_in[2];
  const float* wk = (const float*)d_in[3];
  const float* bk = (const float*)d_in[4];
  const float* wv = (const float*)d_in[5];
  const float* bv = (const float*)d_in[6];
  const float* wl = (const float*)d_in[7];
  const float* bl = (const float*)d_in[8];
  const float* gamma = (const float*)d_in[9];
  float* out = (float*)d_out;

  // workspace carve (f16): wf 512KB | xT 4MB | qT 4MB | kT 4MB | v 4MB | attnT 4MB
  char* ws = (char*)d_ws;
  _Float16* wf    = (_Float16*)(ws);
  _Float16* xT    = (_Float16*)(ws + (512u << 10));
  _Float16* qT    = (_Float16*)(ws + (512u << 10) + 1u * 4194304u);
  _Float16* kT    = (_Float16*)(ws + (512u << 10) + 2u * 4194304u);
  _Float16* vbuf  = (_Float16*)(ws + (512u << 10) + 3u * 4194304u);
  _Float16* attnT = (_Float16*)(ws + (512u << 10) + 4u * 4194304u);

  prep_w_kernel<<<(4 * 65536) / 256, 256, 0, stream>>>(wq, wk, wv, wl, wf);
  prep_x_kernel<<<(2 * CTOT * LTOK) / 256, 256, 0, stream>>>(x, xT);
  proj_kernel<<<1024, 256, 0, stream>>>(xT, wf, bq, bk, bv, qT, kT, vbuf);
  attn_kernel<<<512, 256, 0, stream>>>(kT, qT, vbuf, attnT);
  outproj_kernel<<<1024, 256, 0, stream>>>(attnT, wf + 3 * 65536, bl, x, gamma, out);
}
